// GCN_Model_89438398972170
// MI455X (gfx1250) — matrix operand order repaired and checked by reading, not compiled, not run on hardware
//
#include <hip/hip_runtime.h>
#include <hip/hip_bf16.h>
#include <cstdint>
#include <cstddef>

// ---------------------------------------------------------------------------
// GCN on MI455X (gfx1250, wave32).
// Per layer (3 kernels):
//   1) gemm_wmma : Hlin = Hbf16 @ Wpacked (v_wmma_f32_16x16x32_bf16, f32 accum)
//                  software-pipelined fragment loads (B 2-ahead, A 1-step-ahead)
//                  fused epilogue: acc = Hlin*dis^2 + bias
//   2) edge_scatter: acc[dst] += Hlin[src]*norm   (f32 atomics, L2-resident)
//   3) relu_cvt  : Hnext_bf16 = bf16(max(acc,0))  (layer 3: relu in place, f32)
// Head: segment_max via unsigned atomicMax (post-relu >= 0), matvec + sigmoid.
// Feature widths padded to next layer's K rounded to 32: 75->96, 150->160, 50->64;
// padded columns are exactly 0 (zero-padded packed weights, zero bias).
// ---------------------------------------------------------------------------

typedef __attribute__((ext_vector_type(16))) __bf16 v16bf;
typedef __attribute__((ext_vector_type(8)))  float  v8f;

static constexpr int NN = 50000;
static constexpr int EE = 500000;
static constexpr int GG = 64;

static __device__ __forceinline__ unsigned short f2bf(float f) {
  union { float f; unsigned u; } v; v.f = f;
  unsigned r = v.u + 0x7FFFu + ((v.u >> 16) & 1u);   // round-to-nearest-even
  return (unsigned short)(r >> 16);
}

// ---------------- degree / norm precompute ----------------
__global__ void fill_ones_kernel(float* a, int n) {
  int i = blockIdx.x * 256 + threadIdx.x;
  if (i < n) a[i] = 1.0f;
}
__global__ void deg_kernel(const int* __restrict__ dst, float* __restrict__ deg, int e) {
  int i = blockIdx.x * 256 + threadIdx.x;
  if (i < e) atomicAdd(&deg[dst[i]], 1.0f);
}
__global__ void rsqrt_kernel(float* a, int n) {
  int i = blockIdx.x * 256 + threadIdx.x;
  if (i < n) a[i] = rsqrtf(a[i]);
}
__global__ void norm_kernel(const int* __restrict__ src, const int* __restrict__ dst,
                            const float* __restrict__ dis, float* __restrict__ norm, int e) {
  int i = blockIdx.x * 256 + threadIdx.x;
  if (i < e) norm[i] = dis[src[i]] * dis[dst[i]];
}

// ---------------- f32 -> bf16 cast (4 elems/thread, n % 4 == 0) ----------------
__global__ void cvt_bf16_kernel(const float* __restrict__ in, unsigned short* __restrict__ out, int n) {
  int i = (blockIdx.x * 256 + threadIdx.x) * 4;
  if (i >= n) return;
  float4 f = *reinterpret_cast<const float4*>(in + i);
  ushort4 o;
  o.x = f2bf(f.x); o.y = f2bf(f.y); o.z = f2bf(f.z); o.w = f2bf(f.w);
  *reinterpret_cast<ushort4*>(out + i) = o;
}

// ---------------- relu + f32 -> bf16 cast ----------------
__global__ void relu_cvt_kernel(const float* __restrict__ in, unsigned short* __restrict__ out, int n) {
  int i = (blockIdx.x * 256 + threadIdx.x) * 4;
  if (i >= n) return;
  float4 f = *reinterpret_cast<const float4*>(in + i);
  ushort4 o;
  o.x = f2bf(fmaxf(f.x, 0.0f)); o.y = f2bf(fmaxf(f.y, 0.0f));
  o.z = f2bf(fmaxf(f.z, 0.0f)); o.w = f2bf(fmaxf(f.w, 0.0f));
  *reinterpret_cast<ushort4*>(out + i) = o;
}

__global__ void relu_kernel(float* a, int n) {
  int i = blockIdx.x * 256 + threadIdx.x;
  if (i < n) a[i] = fmaxf(a[i], 0.0f);
}

// ---------------- weight packing into WMMA B-fragment order ----------------
// Fragment element (ks, nt, lane, i) holds W[k][n], k = ks*32 + 16*(i>>3) + 8*(lane>>4) + (i&7),
// n = nt*16 + (lane&15); zero outside (Kreal, Nreal).
template<int KPAD, int BN>
__global__ void pack_w_kernel(const float* __restrict__ W, unsigned short* __restrict__ Bp,
                              int Kreal, int Nreal) {
  constexpr int NT  = BN / 16;
  constexpr int TOT = (KPAD / 32) * NT * 512;
  int idx = blockIdx.x * 256 + threadIdx.x;
  if (idx >= TOT) return;
  int i    = idx & 15;
  int lane = (idx >> 4) & 31;
  int nt   = (idx >> 9) % NT;
  int ks   = idx / (NT * 512);
  int k = ks * 32 + (i >> 3) * 16 + (lane >> 4) * 8 + (i & 7);
  int n = nt * 16 + (lane & 15);
  float v = (k < Kreal && n < Nreal) ? W[k * Nreal + n] : 0.0f;
  Bp[idx] = f2bf(v);
}

// ---------------- WMMA GEMM with fused bias/self-loop epilogue ----------------
// Hlin[M x BN] = A_bf16[M x KPAD] @ Wpacked ; acc = Hlin*dis^2 + bias
template<int KPAD, int BN, int NREAL>
__global__ __launch_bounds__(256)
void gemm_wmma_kernel(const unsigned short* __restrict__ Abf,
                      const unsigned short* __restrict__ Bp,
                      float* __restrict__ Hlin, float* __restrict__ accOut,
                      const float* __restrict__ dis, const float* __restrict__ bias,
                      int M) {
  constexpr int NT = BN / 16;
  constexpr int KS = KPAD / 32;
  constexpr int T  = KS * NT;
  const int wave = threadIdx.x >> 5;
  const int lane = threadIdx.x & 31;
  const int half = lane >> 4;
  const int r16  = lane & 15;
  const int blockM = blockIdx.x * 128 + wave * 16;

  int arow_i = blockM + r16;
  if (arow_i >= M) arow_i = M - 1;                 // clamp: EXEC must stay all-1s for WMMA
  // A fragment for step ks: 2 x uint4 at uint4-offsets {4*ks, 4*ks+2}
  const uint4* abase = reinterpret_cast<const uint4*>(Abf + (size_t)arow_i * KPAD + half * 8);
  // B fragment t = ks*NT + nt: 2 x uint4 at uint4-offsets {64*t, 64*t+1} from lane base
  const uint4* bbase = reinterpret_cast<const uint4*>(Bp) + lane * 2;

  union Frag { v16bf v; uint4 q[2]; };
  Frag af[2], bf[3];
  v8f acc[NT];
  #pragma unroll
  for (int nt = 0; nt < NT; ++nt)
    #pragma unroll
    for (int v = 0; v < 8; ++v) acc[nt][v] = 0.0f;

  // prologue: A(ks=0), B(t=0), B(t=1)
  af[0].q[0] = abase[0];
  af[0].q[1] = abase[2];
  bf[0].q[0] = bbase[0];       bf[0].q[1] = bbase[1];
  bf[1].q[0] = bbase[64];      bf[1].q[1] = bbase[64 + 1];

  #pragma unroll
  for (int t = 0; t < T; ++t) {
    const int ks = t / NT;
    const int nt = t % NT;
    if (t + 2 < T) {                         // B prefetch, 2 WMMAs ahead
      Frag& nb = bf[(t + 2) % 3];
      nb.q[0] = bbase[(t + 2) * 64];
      nb.q[1] = bbase[(t + 2) * 64 + 1];
    }
    if (nt == 0 && ks + 1 < KS) {            // A prefetch, 1 K-step ahead
      Frag& na = af[(ks + 1) & 1];
      na.q[0] = abase[(ks + 1) * 4];
      na.q[1] = abase[(ks + 1) * 4 + 2];
    }
    acc[nt] = __builtin_amdgcn_wmma_f32_16x16x32_bf16(
        false, af[ks & 1].v, false, bf[t % 3].v, (short)0, acc[nt], false, false);
  }

  // bias per lane-column (padded cols -> 0)
  float bcol[NT];
  #pragma unroll
  for (int nt = 0; nt < NT; ++nt) {
    const int col = nt * 16 + r16;
    bcol[nt] = (col < NREAL) ? bias[col] : 0.0f;
  }

  // C/D layout: VGPR v, lane -> M = v + 8*(lane>>4), N = lane&15
  #pragma unroll
  for (int v = 0; v < 8; ++v) {
    const int row = blockM + half * 8 + v;
    if (row < M) {
      const float d  = dis[row];
      const float sn = d * d;
      const size_t base = (size_t)row * BN + r16;
      #pragma unroll
      for (int nt = 0; nt < NT; ++nt) {
        const float h = acc[nt][v];
        Hlin[base + nt * 16]   = h;
        accOut[base + nt * 16] = h * sn + bcol[nt];
      }
    }
  }
}

// ---------------- edge scatter: acc[dst] += Hlin[src]*norm ----------------
template<int F, int BN>
__global__ void edge_scatter_kernel(const int* __restrict__ src, const int* __restrict__ dst,
                                    const float* __restrict__ norm,
                                    const float* __restrict__ Hlin, float* __restrict__ acc) {
  int t = blockIdx.x * 256 + threadIdx.x;
  if (t >= EE * F) return;
  int e = t / F, c = t - e * F;
  float v = Hlin[(size_t)src[e] * BN + c] * norm[e];
  atomicAdd(&acc[(size_t)dst[e] * BN + c], v);
}

// ---------------- pooling head ----------------
__global__ void pool_init_kernel(float* pooled) {
  int i = blockIdx.x * 256 + threadIdx.x;
  if (i < GG * 64) pooled[i] = 0.0f;
}
__global__ void pool_max_kernel(const float* __restrict__ h, const int* __restrict__ batch,
                                float* __restrict__ pooled) {
  int t = blockIdx.x * 256 + threadIdx.x;
  if (t >= NN * 50) return;
  int n = t / 50, c = t - n * 50;
  unsigned v = __float_as_uint(h[(size_t)n * 64 + c]);     // post-relu: >= 0
  atomicMax((unsigned*)&pooled[batch[n] * 64 + c], v);
}
__global__ void head_kernel(const float* __restrict__ pooled, const float* __restrict__ Wo,
                            const float* __restrict__ bo, float* __restrict__ out) {
  int g = threadIdx.x;
  if (g >= GG) return;
  float s = bo[0];
  #pragma unroll
  for (int c = 0; c < 50; ++c) s += pooled[g * 64 + c] * Wo[c];
  out[g] = 1.0f / (1.0f + expf(-s));
}

// ---------------------------------------------------------------------------
extern "C" void kernel_launch(void* const* d_in, const int* in_sizes, int n_in,
                              void* d_out, int out_size, void* d_ws, size_t ws_size,
                              hipStream_t stream) {
  (void)in_sizes; (void)n_in; (void)out_size; (void)ws_size;
  const float* x   = (const float*)d_in[0];
  const int*   ei  = (const int*)d_in[1];
  const int*   src = ei, *dst = ei + EE;
  const int*   bat = (const int*)d_in[2];
  const float* W1 = (const float*)d_in[3], *b1 = (const float*)d_in[4];
  const float* W2 = (const float*)d_in[5], *b2 = (const float*)d_in[6];
  const float* W3 = (const float*)d_in[7], *b3 = (const float*)d_in[8];
  const float* Wo = (const float*)d_in[9], *bo = (const float*)d_in[10];
  float* out = (float*)d_out;

  char* ws = (char*)d_ws;
  size_t off = 0;
  auto alloc = [&](size_t bytes) -> void* {
    void* p = ws + off;
    off = (off + bytes + 255) & ~(size_t)255;
    return p;
  };
  float*          dis    = (float*)alloc((size_t)NN * 4);
  float*          norm   = (float*)alloc((size_t)EE * 4);
  unsigned short* Bp1    = (unsigned short*)alloc((size_t)(512 / 32) * 6  * 512 * 2);
  unsigned short* Bp2    = (unsigned short*)alloc((size_t)(96  / 32) * 10 * 512 * 2);
  unsigned short* Bp3    = (unsigned short*)alloc((size_t)(160 / 32) * 4  * 512 * 2);
  float*          pooled = (float*)alloc((size_t)GG * 64 * 4);
  unsigned short* xb     = (unsigned short*)alloc((size_t)NN * 512 * 2);  // bf16 x
  unsigned short* h1b    = (unsigned short*)alloc((size_t)NN * 96  * 2);  // bf16 layer-1 out
  unsigned short* h2b    = (unsigned short*)alloc((size_t)NN * 160 * 2);  // bf16 layer-2 out
  float*          Hlin   = (float*)alloc((size_t)NN * 160 * 4);           // GEMM output (f32)
  float*          acc    = (float*)alloc((size_t)NN * 160 * 4);           // aggregation acc (f32)

  const int TB = 256;
  auto cdiv = [](long long a, long long b) { return (int)((a + b - 1) / b); };
  const int gridM = cdiv(NN, 128);

  // degrees & norms
  fill_ones_kernel<<<cdiv(NN, TB), TB, 0, stream>>>(dis, NN);
  deg_kernel<<<cdiv(EE, TB), TB, 0, stream>>>(dst, dis, EE);
  rsqrt_kernel<<<cdiv(NN, TB), TB, 0, stream>>>(dis, NN);
  norm_kernel<<<cdiv(EE, TB), TB, 0, stream>>>(src, dst, dis, norm, EE);

  // pack weights into WMMA fragment order; cast x to bf16
  pack_w_kernel<512, 96><<<cdiv((512/32)*6*512,  TB), TB, 0, stream>>>(W1, Bp1, 512, 75);
  pack_w_kernel<96, 160><<<cdiv((96/32)*10*512,  TB), TB, 0, stream>>>(W2, Bp2, 75, 150);
  pack_w_kernel<160, 64><<<cdiv((160/32)*4*512,  TB), TB, 0, stream>>>(W3, Bp3, 150, 50);
  cvt_bf16_kernel<<<cdiv((long long)NN * 512 / 4, TB), TB, 0, stream>>>(x, xb, NN * 512);

  // ---- layer 1: xb[50000x512] -> acc[50000x96] -> h1b ----
  gemm_wmma_kernel<512, 96, 75><<<gridM, TB, 0, stream>>>(xb, Bp1, Hlin, acc, dis, b1, NN);
  edge_scatter_kernel<75, 96><<<cdiv((long long)EE * 75, TB), TB, 0, stream>>>(src, dst, norm, Hlin, acc);
  relu_cvt_kernel<<<cdiv((long long)NN * 96 / 4, TB), TB, 0, stream>>>(acc, h1b, NN * 96);

  // ---- layer 2: h1b[50000x96] -> acc[50000x160] -> h2b ----
  gemm_wmma_kernel<96, 160, 150><<<gridM, TB, 0, stream>>>(h1b, Bp2, Hlin, acc, dis, b2, NN);
  edge_scatter_kernel<150, 160><<<cdiv((long long)EE * 150, TB), TB, 0, stream>>>(src, dst, norm, Hlin, acc);
  relu_cvt_kernel<<<cdiv((long long)NN * 160 / 4, TB), TB, 0, stream>>>(acc, h2b, NN * 160);

  // ---- layer 3: h2b[50000x160] -> acc[50000x64] (relu in place, f32) ----
  gemm_wmma_kernel<160, 64, 50><<<gridM, TB, 0, stream>>>(h2b, Bp3, Hlin, acc, dis, b3, NN);
  edge_scatter_kernel<50, 64><<<cdiv((long long)EE * 50, TB), TB, 0, stream>>>(src, dst, norm, Hlin, acc);
  relu_kernel<<<cdiv((long long)NN * 64, TB), TB, 0, stream>>>(acc, NN * 64);

  // ---- pooling + head ----
  pool_init_kernel<<<cdiv(GG * 64, TB), TB, 0, stream>>>(pooled);
  pool_max_kernel<<<cdiv((long long)NN * 50, TB), TB, 0, stream>>>(acc, bat, pooled);
  head_kernel<<<1, 64, 0, stream>>>(pooled, Wo, bo, out);
}
